// GraphRecurrentNeuralNetwork_81363860455700
// MI455X (gfx1250) — matrix operand order repaired and checked by reading, not compiled, not run on hardware
//
#include <hip/hip_runtime.h>
#include <hip/hip_bf16.h>

// ---------------- types for WMMA fragments ----------------
typedef __attribute__((ext_vector_type(16))) __bf16 v16bf;
typedef __attribute__((ext_vector_type(8)))  float  v8f;

union Frag {
    unsigned int u[8];
    v16bf v;
};

__device__ __forceinline__ unsigned short f2bf(float f) {
    unsigned int u = __float_as_uint(f);
    unsigned int r = u + 0x7FFFu + ((u >> 16) & 1u);   // round-to-nearest-even
    return (unsigned short)(r >> 16);
}

// ---------------- constants ----------------
#define NN   10000
#define DD   256
#define KTOT 1024          // concat(aggX, X, aggH, h)

// ---------------- kernel 1: in-degree ----------------
__global__ void deg_kernel(const long long* __restrict__ ei, float* __restrict__ deg, int E) {
    int i = blockIdx.x * blockDim.x + threadIdx.x;
    if (i < E) {
        int dst = (int)ei[(size_t)E + i];
        atomicAdd(&deg[dst], 1.0f);
    }
}

// ---------------- kernel 2: scatter-add of X[src], h[src] into dst rows ------
// One 256-thread block per edge; aggX/aggH (~20MB) stay L2-resident on MI455X.
__global__ void scatter_kernel(const long long* __restrict__ ei,
                               const float* __restrict__ X, const float* __restrict__ h,
                               float* __restrict__ aggX, float* __restrict__ aggH, int E) {
    int e = blockIdx.x;
    int src = (int)ei[e];
    int dst = (int)ei[(size_t)E + e];
    int c = threadIdx.x;
    atomicAdd(&aggX[(size_t)dst * DD + c], X[(size_t)src * DD + c]);
    atomicAdd(&aggH[(size_t)dst * DD + c], h[(size_t)src * DD + c]);
}

// ---------------- kernel 3: normalize + pack bf16 A matrix [N, 1024] ---------
__global__ void build_A_kernel(const float* __restrict__ X, const float* __restrict__ h,
                               const float* __restrict__ aggX, const float* __restrict__ aggH,
                               const float* __restrict__ deg, unsigned short* __restrict__ Abf) {
    int n = blockIdx.x;
    int c = threadIdx.x;
    float d = deg[n];
    float inv = 1.0f / (d > 1.0f ? d : 1.0f);
    size_t o = (size_t)n * KTOT;
    size_t p = (size_t)n * DD + c;
    Abf[o + c]           = f2bf(aggX[p] * inv);
    Abf[o + 256 + c]     = f2bf(X[p]);
    Abf[o + 512 + c]     = f2bf(aggH[p] * inv);
    Abf[o + 768 + c]     = f2bf(h[p]);
}

// ---------------- kernel 4: pack bf16 weight matrix, transposed W[j][k] ------
// j = gate*256 + out_channel (0..1023), k = concat input (0..1023)
__global__ void build_W_kernel(const float* __restrict__ Wx_l, const float* __restrict__ Wx_r,
                               const float* __restrict__ Wh_l, const float* __restrict__ Wh_r,
                               unsigned short* __restrict__ Wbf) {
    int idx = blockIdx.x * blockDim.x + threadIdx.x;   // 0 .. 1M-1
    int j = idx >> 10;
    int k = idx & 1023;
    int g = j >> 8, e = j & 255;
    int sel = k >> 8, kin = k & 255;
    const float* W = (sel == 0) ? Wx_l : (sel == 1) ? Wx_r : (sel == 2) ? Wh_l : Wh_r;
    float v = W[((size_t)g << 16) + ((size_t)kin << 8) + e];
    Wbf[(size_t)j * KTOT + k] = f2bf(v);
}

// ---------------- kernel 5: fused WMMA GEMM + LSTM gate epilogue -------------
// Block: 256 threads = 8 wave32 waves; tile = 16 nodes x 1024 gate outputs.
// Wave w owns 8 output tiles (columns w*128 .. w*128+127).
// All 8 B fragments are preloaded per K-step so the 8 WMMAs issue back-to-back
// (single load clause + one wait), making the A operand reuse hint legal.
__global__ __launch_bounds__(256) void gemm_gates_kernel(
        const unsigned short* __restrict__ Abf, const unsigned short* __restrict__ Wbf,
        const float* __restrict__ c_in,
        const float* __restrict__ bx_l, const float* __restrict__ bh_l,
        const float* __restrict__ wc, const float* __restrict__ bb,
        float* __restrict__ outO, float* __restrict__ outH, float* __restrict__ outC) {
    const int S = 1032;                       // LDS row stride (floats), padded
    extern __shared__ float lds[];            // 16 * 1032 * 4 = 66,048 B

    int tid  = threadIdx.x;
    int wave = tid >> 5;
    int lane = tid & 31;
    int half = lane >> 4;
    int l16  = lane & 15;
    int n0   = blockIdx.x << 4;

    const unsigned int* Arow = (const unsigned int*)Abf + (size_t)(n0 + l16) * (KTOT / 2);
    const unsigned int* Brow[8];
#pragma unroll
    for (int t = 0; t < 8; ++t) {
        int col = wave * 128 + t * 16 + l16;
        Brow[t] = (const unsigned int*)Wbf + (size_t)col * (KTOT / 2);
    }

    v8f acc[8];
#pragma unroll
    for (int t = 0; t < 8; ++t)
        acc[t] = (v8f){0.f, 0.f, 0.f, 0.f, 0.f, 0.f, 0.f, 0.f};

#pragma unroll 1
    for (int kk = 0; kk < KTOT; kk += 32) {
        // A fragment: ISA 16-bit A 16x32 layout (lane M=l16; K pairs per dword)
        Frag a;
#pragma unroll
        for (int v = 0; v < 8; ++v) {
            int ka = kk + ((v < 4) ? (2 * v + 8 * half) : (16 + 2 * (v - 4) + 8 * half));
            a.u[v] = Arow[ka >> 1];
        }
        // Preload ALL eight B fragments -> one load clause, one wait, 8 WMMAs
        Frag b[8];
#pragma unroll
        for (int t = 0; t < 8; ++t) {
#pragma unroll
            for (int v = 0; v < 8; ++v) {
                int kb = kk + ((v < 4) ? (2 * v + 8 * half) : (16 + 2 * (v - 4) + 8 * half));   // B: same K pairs as A (table order)
                b[t].u[v] = Brow[t][kb >> 1];
            }
        }
        // Pull next K-block of A toward the WGP cache while WMMAs run
        if (kk + 32 < KTOT)
            __builtin_prefetch((const void*)(Arow + ((kk + 32) >> 1)), 0, 3);

        // reuse_a must be a frontend integer constant -> explicit unroll by macro.
        // WMMAs 1..7 read the same A VGPRs as the immediately preceding WMMA.
#define WMMA_STEP(T, RA)                                                        \
        acc[T] = __builtin_amdgcn_wmma_f32_16x16x32_bf16(                       \
            false, a.v, false, b[T].v, (short)0, acc[T], RA, false)
        WMMA_STEP(0, false);
        WMMA_STEP(1, true);
        WMMA_STEP(2, true);
        WMMA_STEP(3, true);
        WMMA_STEP(4, true);
        WMMA_STEP(5, true);
        WMMA_STEP(6, true);
        WMMA_STEP(7, true);
#undef WMMA_STEP
    }

    // D layout: lane holds (M = v + 8*half, Ncol = l16) per acc dword -> LDS exchange
#pragma unroll
    for (int t = 0; t < 8; ++t) {
        int j = wave * 128 + t * 16 + l16;
#pragma unroll
        for (int v = 0; v < 8; ++v) {
            int m = v + 8 * half;
            lds[m * S + j] = acc[t][v];
        }
    }
    __syncthreads();

    // Epilogue: thread tid = channel e, all 16 rows of the block tile
    int e = tid;
    float wc0 = wc[e], wc1 = wc[256 + e], wc2 = wc[512 + e];
    float bi  = bx_l[e]       + bh_l[e]       + bb[e];
    float bf_ = bx_l[256 + e] + bh_l[256 + e] + bb[256 + e];
    float bc  = bx_l[512 + e] + bh_l[512 + e] + bb[512 + e];
    float bo  = bx_l[768 + e] + bh_l[768 + e] + bb[768 + e];
#pragma unroll 1
    for (int i = 0; i < 16; ++i) {
        int n = n0 + i;
        float cv = c_in[(size_t)n * DD + e];
        float pi = lds[i * S + e]       + bi  + wc0 * cv;
        float pf = lds[i * S + 256 + e] + bf_ + wc1 * cv;
        float pc = lds[i * S + 512 + e] + bc;
        float po = lds[i * S + 768 + e] + bo;
        float I  = 1.0f / (1.0f + __expf(-pi));
        float F  = 1.0f / (1.0f + __expf(-pf));
        float T  = tanhf(pc);
        float Cn = F * cv + I * T;
        float O  = 1.0f / (1.0f + __expf(-(po + wc2 * Cn)));
        float Hn = O * tanhf(Cn);
        size_t o = (size_t)n * DD + e;
        outO[o] = O;
        outH[o] = Hn;
        outC[o] = Cn;
    }
}

// ---------------- host-side launch ----------------
extern "C" void kernel_launch(void* const* d_in, const int* in_sizes, int n_in,
                              void* d_out, int out_size, void* d_ws, size_t ws_size,
                              hipStream_t stream) {
    const float*     X    = (const float*)d_in[0];
    const long long* ei   = (const long long*)d_in[1];   // int64 (2, E)
    const float*     h    = (const float*)d_in[2];
    const float*     c    = (const float*)d_in[3];
    const float*     Wx_l = (const float*)d_in[4];
    const float*     bx_l = (const float*)d_in[5];
    const float*     Wx_r = (const float*)d_in[6];
    const float*     Wh_l = (const float*)d_in[7];
    const float*     bh_l = (const float*)d_in[8];
    const float*     Wh_r = (const float*)d_in[9];
    const float*     wc   = (const float*)d_in[10];
    const float*     bb   = (const float*)d_in[11];

    const int E = in_sizes[1] / 2;

    // workspace layout
    char* ws = (char*)d_ws;
    float*          deg  = (float*)ws;                                     // 40 KB (padded)
    float*          aggX = (float*)(ws + 40960);                           // N*D f32
    float*          aggH = (float*)(ws + 40960 + (size_t)NN * DD * 4);     // N*D f32
    unsigned short* Abf  = (unsigned short*)(ws + 40960 + 2 * (size_t)NN * DD * 4);
    unsigned short* Wbf  = (unsigned short*)(ws + 40960 + 2 * (size_t)NN * DD * 4
                                                + (size_t)NN * KTOT * 2);

    // zero the accumulation buffers (workspace is poisoned)
    (void)hipMemsetAsync(deg, 0, 40960 + 2 * (size_t)NN * DD * 4, stream);

    deg_kernel    <<<(E + 255) / 256, 256, 0, stream>>>(ei, deg, E);
    scatter_kernel<<<E, 256, 0, stream>>>(ei, X, h, aggX, aggH, E);
    build_A_kernel<<<NN, 256, 0, stream>>>(X, h, aggX, aggH, deg, Abf);
    build_W_kernel<<<(KTOT * KTOT) / 256, 256, 0, stream>>>(Wx_l, Wx_r, Wh_l, Wh_r, Wbf);

    float* outO = (float*)d_out;
    float* outH = outO + (size_t)NN * DD;
    float* outC = outO + 2 * (size_t)NN * DD;
    gemm_gates_kernel<<<NN / 16, 256, 16 * 1032 * sizeof(float), stream>>>(
        Abf, Wbf, c, bx_l, bh_l, wc, bb, outO, outH, outC);
}
